// CVRAE_next_action_3693671874561
// MI455X (gfx1250) — matrix operand order repaired and checked by reading, not compiled, not run on hardware
//
#include <hip/hip_runtime.h>
#include <math.h>

#define Bsz 256
#define Ssz 512
#define Asz 64
#define Hsz 1024
#define Zsz 256

typedef unsigned short u16;
typedef __attribute__((ext_vector_type(16))) __bf16 v16bf;
typedef __attribute__((ext_vector_type(8)))  __bf16 v8bf;
typedef __attribute__((ext_vector_type(8)))  float  v8f;

__device__ __forceinline__ u16 f2bf(float f) {
  unsigned u = __builtin_bit_cast(unsigned, f);
  u += 0x7FFFu + ((u >> 16) & 1u);          // round-to-nearest-even
  return (u16)(u >> 16);
}
__device__ __forceinline__ float softplusf(float x){ return x > 20.f ? x : log1pf(expf(x)); }
__device__ __forceinline__ float sigmf(float x){ return 1.f/(1.f+expf(-x)); }

// Generic pointer to a __shared__ object: low 32 bits of the flat address are
// the wave-relative LDS offset (ISA 10.2).
__device__ __forceinline__ unsigned lds_off(const void* p) {
  return (unsigned)(unsigned long long)p;
}

// CDNA5 async copy global -> LDS, 16B per lane, tracked by ASYNCcnt.
// GVS mode: mem = SADDR(64) + VADDR(32) ; VDST holds LDS byte offset.
__device__ __forceinline__ void async_ld_b128(unsigned ldsdst, unsigned goff,
                                              const void* sbase) {
  asm volatile("global_load_async_to_lds_b128 %0, %1, %2"
               :: "v"(ldsdst), "v"(goff), "s"(sbase) : "memory");
}
#define WAIT_ASYNC(n) asm volatile("s_wait_asynccnt " #n ::: "memory")

// ---------------------------------------------------------------------------
// Tiled bf16 WMMA GEMM:  C[M,N] = A[M,K] @ W[N,K]^T + bias.
// OUTBF=1 -> bf16 output with relu (all hidden layers); OUTBF=0 -> fp32, no act.
// Block = 128 threads = 4 waves (2x2), block tile 128x64, wave tile 64x32,
// K-step 32, double-buffered LDS staged with global_load_async_to_lds_b128.
// Per K-step per wave: 8 v_wmma against 12 ds_load_b128 (A regs reused over N).
// ---------------------------------------------------------------------------
#define BM 128
#define BN 64
#define BKC 32
#define LDSP 40           // padded row (elems): 80B = 20 banks -> conflict-free b128
#define STAGE_A (BM*LDSP) // elems per A stage
#define STAGE_B (BN*LDSP)

template <bool OUTBF>
__global__ __launch_bounds__(128)
void gemm_bf16_wmma(const u16* __restrict__ Ap, const u16* __restrict__ Wp,
                    const float* __restrict__ bias,
                    float* __restrict__ outF, u16* __restrict__ outB,
                    int M, int N, int K, int ldc, int coff)
{
  __shared__ __align__(16) u16 sA[2][STAGE_A];
  __shared__ __align__(16) u16 sB[2][STAGE_B];

  const int tid  = threadIdx.x;
  const int lane = tid & 31;
  const int wave = tid >> 5;
  const int half = lane >> 4;
  const int l16  = lane & 15;

  const int m0 = blockIdx.y * BM;
  const int n0 = blockIdx.x * BN;
  const int wm = (wave >> 1) * 64;   // wave tile: 64 rows
  const int wn = (wave & 1) * 32;    //            32 cols

  // staging map: idx = tid + it*128 ; row = idx>>2 ; col8 = (idx&3)*8 elems
  const int rr = tid >> 2, cc = (tid & 3) << 3;
  unsigned lA[4], gA[4], lB[2], gB[2];
#pragma unroll
  for (int it = 0; it < 4; ++it) {   // A tile: 128 rows x 32 elems
    int r = rr + it * 32;
    lA[it] = lds_off(&sA[0][0]) + (unsigned)(r * LDSP + cc) * 2u;
    gA[it] = ((unsigned)(m0 + r) * (unsigned)K + (unsigned)cc) * 2u;
  }
#pragma unroll
  for (int it = 0; it < 2; ++it) {   // B tile: 64 rows x 32 elems
    int r = rr + it * 32;
    lB[it] = lds_off(&sB[0][0]) + (unsigned)(r * LDSP + cc) * 2u;
    gB[it] = ((unsigned)(n0 + r) * (unsigned)K + (unsigned)cc) * 2u;
  }

  auto issue = [&](int k0, int buf) {
    const unsigned kb = (unsigned)k0 * 2u;
    const unsigned oA = buf ? (unsigned)STAGE_A * 2u : 0u;
    const unsigned oB = buf ? (unsigned)STAGE_B * 2u : 0u;
#pragma unroll
    for (int it = 0; it < 4; ++it) async_ld_b128(lA[it] + oA, gA[it] + kb, Ap);
#pragma unroll
    for (int it = 0; it < 2; ++it) async_ld_b128(lB[it] + oB, gB[it] + kb, Wp);
  };

  v8f acc[4][2];
#pragma unroll
  for (int i = 0; i < 4; ++i)
#pragma unroll
    for (int j = 0; j < 2; ++j) { v8f z = {0,0,0,0,0,0,0,0}; acc[i][j] = z; }

  const int nst = K / BKC;
  issue(0, 0);

  for (int s = 0; s < nst; ++s) {
    const int buf = s & 1;
    if (s + 1 < nst) {
      issue((s + 1) * BKC, buf ^ 1);   // prefetch next stage into other buffer
      WAIT_ASYNC(6);                   // oldest 6 (this stage) complete, in order
    } else {
      WAIT_ASYNC(0);
    }
    __syncthreads();                   // everyone's stage-s data visible in LDS

    const u16* sAb = &sA[buf][0];
    const u16* sBb = &sB[buf][0];

    // A operand: row = m, elems [0..7]=K(8*half..), [8..15]=K(16+8*half..)  (ISA 7.12.2)
    v16bf av[4], bv[2];
#pragma unroll
    for (int mi = 0; mi < 4; ++mi) {
      const u16* p = &sAb[(wm + mi*16 + l16) * LDSP];
      v8bf lo = *(const v8bf*)(p + 8*half);
      v8bf hi = *(const v8bf*)(p + 16 + 8*half);
#pragma unroll
      for (int e = 0; e < 8; ++e) { av[mi][e] = lo[e]; av[mi][e+8] = hi[e]; }
    }
    // B operand: col = n in lanes, elems [0..7]=K(8*half..), [8..15]=K(16+8*half..)
#pragma unroll
    for (int ni = 0; ni < 2; ++ni) {
      const u16* p = &sBb[(wn + ni*16 + l16) * LDSP];
      v8bf lo = *(const v8bf*)(p + 8*half);
      v8bf hi = *(const v8bf*)(p + 16 + 8*half);
#pragma unroll
      for (int e = 0; e < 8; ++e) { bv[ni][e] = lo[e]; bv[ni][e+8] = hi[e]; }
    }

#pragma unroll
    for (int mi = 0; mi < 4; ++mi)
#pragma unroll
      for (int ni = 0; ni < 2; ++ni)
        acc[mi][ni] = __builtin_amdgcn_wmma_f32_16x16x32_bf16(
            false, av[mi], false, bv[ni], (short)0, acc[mi][ni], false, false);

    __syncthreads();                   // all waves done reading buf before reuse
  }

  // epilogue: D layout m = r + 8*half, n = l16
#pragma unroll
  for (int mi = 0; mi < 4; ++mi)
#pragma unroll
    for (int ni = 0; ni < 2; ++ni) {
      int n  = n0 + wn + ni*16 + l16;
      int mb = m0 + wm + mi*16 + 8*half;
      float bvv = bias[n];
#pragma unroll
      for (int r = 0; r < 8; ++r) {
        float v = acc[mi][ni][r] + bvv;
        size_t off = (size_t)(mb + r) * ldc + coff + n;
        if (OUTBF) outB[off] = f2bf(fmaxf(v, 0.f));
        else       outF[off] = v;
      }
    }
}

// ---------------------------------------------------------------------------
// Elementwise kernels
// ---------------------------------------------------------------------------
__global__ void cvt_f32_bf16(const float* __restrict__ src, u16* __restrict__ dst, int n) {
  int i = blockIdx.x * 256 + threadIdx.x;
  if (i < n) dst[i] = f2bf(src[i]);
}

__global__ void init_h(const float* __restrict__ h0, float* __restrict__ h,
                       u16* __restrict__ h_bf, u16* __restrict__ xh, float* __restrict__ kld) {
  int i = blockIdx.x * 256 + threadIdx.x;          // over B*H
  float v = h0[i];
  h[i] = v;
  u16 b = f2bf(v);
  h_bf[i] = b;
  int bb = i / Hsz, j = i % Hsz;
  xh[(size_t)bb*2*Hsz + Hsz + j] = b;              // [px,h] high half
  if (i == 0) *kld = 0.f;
}

// px = relu(one_hot(x_t) @ phi_x_w.T + b) == relu(phi_x_w[:, x_t] + b)  (gather)
__global__ void embed_px(const int* __restrict__ x, const float* __restrict__ w,
                         const float* __restrict__ bias, u16* __restrict__ xh,
                         u16* __restrict__ xz, int t) {
  int i = blockIdx.x * 256 + threadIdx.x;          // over B*H
  int bb = i / Hsz, j = i % Hsz;
  int a = x[bb*Ssz + t];
  float v = fmaxf(w[j*Asz + a] + bias[j], 0.f);
  u16 bv = f2bf(v);
  size_t o = (size_t)bb*2*Hsz + j;
  xh[o] = bv;                                      // [px,h] low half
  xz[o] = bv;                                      // [px,pz] low half
}

// softplus, KLD (block reduce + atomic), reparam z
__global__ void latent_k(const float* __restrict__ enc_f, const float* __restrict__ pri_f,
                         const float* __restrict__ eps_t, u16* __restrict__ z_bf,
                         float* __restrict__ kld, int use_prior, int add_kld) {
  int i = blockIdx.x * 256 + threadIdx.x;          // over B*Z
  float em = enc_f[i], pm = pri_f[i];
  float es = softplusf(em), ps = softplusf(pm);
  float e  = eps_t[i];
  float zv = use_prior ? (e * ps + pm) : (e * es + em);
  z_bf[i] = f2bf(zv);
  if (add_kld) {
    float d   = em - pm;
    float kel = 2.f*(logf(ps) - logf(es)) + (es*es + d*d) / (ps*ps) - 1.f;
    __shared__ float red[256];
    red[threadIdx.x] = kel;
    __syncthreads();
    for (int st = 128; st > 0; st >>= 1) {
      if (threadIdx.x < st) red[threadIdx.x] += red[threadIdx.x + st];
      __syncthreads();
    }
    if (threadIdx.x == 0) atomicAdd(kld, 0.5f * red[0]);
  }
}

// PyTorch GRU gate math; writes h (fp32), h_bf, and the [px,h] high half for next step
__global__ void gru_gate(const float* __restrict__ gi, const float* __restrict__ gh,
                         float* __restrict__ h, u16* __restrict__ h_bf, u16* __restrict__ xh) {
  int i = blockIdx.x * 256 + threadIdx.x;          // over B*H
  int bb = i / Hsz, j = i % Hsz;
  size_t base = (size_t)bb * 3 * Hsz;
  float r  = sigmf(gi[base + j]         + gh[base + j]);
  float zg = sigmf(gi[base + Hsz + j]   + gh[base + Hsz + j]);
  float n  = tanhf(gi[base + 2*Hsz + j] + r * gh[base + 2*Hsz + j]);
  float hn = (1.f - zg) * n + zg * h[i];
  h[i] = hn;
  u16 b = f2bf(hn);
  h_bf[i] = b;
  xh[(size_t)bb*2*Hsz + Hsz + j] = b;
}

__global__ void copy_h_to_dech(const u16* __restrict__ h_bf, u16* __restrict__ dech) {
  int i = blockIdx.x * 256 + threadIdx.x;          // over B*H
  int bb = i / Hsz, j = i % Hsz;
  dech[(size_t)bb*2*Hsz + Hsz + j] = h_bf[i];
}

__global__ void write_kld(const float* __restrict__ kld, float* __restrict__ out) {
  out[Bsz * Asz] = *kld;
}

// ---------------------------------------------------------------------------
extern "C" void kernel_launch(void* const* d_in, const int* in_sizes, int n_in,
                              void* d_out, int out_size, void* d_ws, size_t ws_size,
                              hipStream_t stream) {
  (void)in_sizes; (void)n_in; (void)out_size; (void)ws_size;

  const int*   x       = (const int*)  d_in[0];
  const float* h0      = (const float*)d_in[1];
  const float* eps     = (const float*)d_in[2];
  const float* phi_x_w = (const float*)d_in[3];
  const float* phi_x_b = (const float*)d_in[4];
  const float* enc_w1  = (const float*)d_in[5];
  const float* enc_b1  = (const float*)d_in[6];
  const float* enc_w2  = (const float*)d_in[7];
  const float* enc_b2  = (const float*)d_in[8];
  const float* pri_w1  = (const float*)d_in[9];
  const float* pri_b1  = (const float*)d_in[10];
  const float* pri_w2  = (const float*)d_in[11];
  const float* pri_b2  = (const float*)d_in[12];
  const float* phi_z_w = (const float*)d_in[13];
  const float* phi_z_b = (const float*)d_in[14];
  const float* dec_w   = (const float*)d_in[15];
  const float* dec_b   = (const float*)d_in[16];
  const float* act_w   = (const float*)d_in[17];
  const float* act_b   = (const float*)d_in[18];
  const float* gru_wih = (const float*)d_in[19];
  const float* gru_whh = (const float*)d_in[20];
  const float* gru_bih = (const float*)d_in[21];
  const float* gru_bhh = (const float*)d_in[22];

  char* ws = (char*)d_ws;
  size_t cur = 0;
  auto alloc = [&](size_t bytes) -> char* {
    char* p = ws + cur;
    cur += (bytes + 255) & ~(size_t)255;
    return p;
  };

  // bf16 weight copies (resident in L2 across the whole sequence)
  u16* enc_w1_b  = (u16*)alloc((size_t)Hsz*2*Hsz*2);
  u16* enc_w2_b  = (u16*)alloc((size_t)Zsz*Hsz*2);
  u16* pri_w1_b  = (u16*)alloc((size_t)Hsz*Hsz*2);
  u16* pri_w2_b  = (u16*)alloc((size_t)Zsz*Hsz*2);
  u16* phi_z_w_b = (u16*)alloc((size_t)Hsz*Zsz*2);
  u16* dec_w_b   = (u16*)alloc((size_t)Hsz*2*Hsz*2);
  u16* act_w_b   = (u16*)alloc((size_t)Asz*Hsz*2);
  u16* gru_wih_b = (u16*)alloc((size_t)3*Hsz*2*Hsz*2);
  u16* gru_whh_b = (u16*)alloc((size_t)3*Hsz*Hsz*2);
  // activations
  float* h     = (float*)alloc((size_t)Bsz*Hsz*4);
  float* gi    = (float*)alloc((size_t)Bsz*3*Hsz*4);
  float* gh    = (float*)alloc((size_t)Bsz*3*Hsz*4);
  float* enc_f = (float*)alloc((size_t)Bsz*Zsz*4);
  float* pri_f = (float*)alloc((size_t)Bsz*Zsz*4);
  float* kld   = (float*)alloc(256);
  u16* xh       = (u16*)alloc((size_t)Bsz*2*Hsz*2);   // [px, h]
  u16* xz       = (u16*)alloc((size_t)Bsz*2*Hsz*2);   // [px, pz]
  u16* dech     = (u16*)alloc((size_t)Bsz*2*Hsz*2);   // [pz, h]
  u16* h_bf     = (u16*)alloc((size_t)Bsz*Hsz*2);
  u16* enc_h_bf = (u16*)alloc((size_t)Bsz*Hsz*2);
  u16* pri_h_bf = (u16*)alloc((size_t)Bsz*Hsz*2);
  u16* dec_bf   = (u16*)alloc((size_t)Bsz*Hsz*2);
  u16* z_bf     = (u16*)alloc((size_t)Bsz*Zsz*2);

  auto cvt = [&](const float* src, u16* dst, int n) {
    cvt_f32_bf16<<<(n + 255) / 256, 256, 0, stream>>>(src, dst, n);
  };
  cvt(enc_w1,  enc_w1_b,  Hsz*2*Hsz);
  cvt(enc_w2,  enc_w2_b,  Zsz*Hsz);
  cvt(pri_w1,  pri_w1_b,  Hsz*Hsz);
  cvt(pri_w2,  pri_w2_b,  Zsz*Hsz);
  cvt(phi_z_w, phi_z_w_b, Hsz*Zsz);
  cvt(dec_w,   dec_w_b,   Hsz*2*Hsz);
  cvt(act_w,   act_w_b,   Asz*Hsz);
  cvt(gru_wih, gru_wih_b, 3*Hsz*2*Hsz);
  cvt(gru_whh, gru_whh_b, 3*Hsz*Hsz);

  auto gemm = [&](const u16* Ab, const u16* Wb, const float* bias,
                  float* outF, u16* outB, int M, int N, int K, int ldc, int coff) {
    dim3 g(N / BN, M / BM);
    if (outB)
      gemm_bf16_wmma<true ><<<g, 128, 0, stream>>>(Ab, Wb, bias, nullptr, outB, M, N, K, ldc, coff);
    else
      gemm_bf16_wmma<false><<<g, 128, 0, stream>>>(Ab, Wb, bias, outF, nullptr, M, N, K, ldc, coff);
  };

  const int BH = Bsz * Hsz;   // 262144
  const int BZ = Bsz * Zsz;   // 65536

  init_h<<<BH / 256, 256, 0, stream>>>(h0, h, h_bf, xh, kld);

  for (int t = 0; t < Ssz; ++t) {
    embed_px<<<BH / 256, 256, 0, stream>>>(x, phi_x_w, phi_x_b, xh, xz, t);
    // enc_h = relu([px,h] @ enc_w1.T + b1)
    gemm(xh, enc_w1_b, enc_b1, nullptr, enc_h_bf, Bsz, Hsz, 2*Hsz, Hsz, 0);
    // pri_h = relu(h @ pri_w1.T + b1)
    gemm(h_bf, pri_w1_b, pri_b1, nullptr, pri_h_bf, Bsz, Hsz, Hsz, Hsz, 0);
    // enc = enc_h @ enc_w2.T + b2 ; pri = pri_h @ pri_w2.T + b2   (fp32)
    gemm(enc_h_bf, enc_w2_b, enc_b2, enc_f, nullptr, Bsz, Zsz, Hsz, Zsz, 0);
    gemm(pri_h_bf, pri_w2_b, pri_b2, pri_f, nullptr, Bsz, Zsz, Hsz, Zsz, 0);
    latent_k<<<BZ / 256, 256, 0, stream>>>(enc_f, pri_f, eps + (size_t)t*Bsz*Zsz,
                                           z_bf, kld, 0, 1);
    // pz = relu(z @ phi_z_w.T + b) -> directly into [px,pz] high half
    gemm(z_bf, phi_z_w_b, phi_z_b, nullptr, xz, Bsz, Hsz, Zsz, 2*Hsz, Hsz);
    // GRU gate pre-activations
    gemm(xz,   gru_wih_b, gru_bih, gi, nullptr, Bsz, 3*Hsz, 2*Hsz, 3*Hsz, 0);
    gemm(h_bf, gru_whh_b, gru_bhh, gh, nullptr, Bsz, 3*Hsz, Hsz,   3*Hsz, 0);
    gru_gate<<<BH / 256, 256, 0, stream>>>(gi, gh, h, h_bf, xh);
  }

  // final prediction head
  gemm(h_bf, pri_w1_b, pri_b1, nullptr, pri_h_bf, Bsz, Hsz, Hsz, Hsz, 0);
  gemm(pri_h_bf, pri_w2_b, pri_b2, pri_f, nullptr, Bsz, Zsz, Hsz, Zsz, 0);
  latent_k<<<BZ / 256, 256, 0, stream>>>(pri_f, pri_f, eps + (size_t)Ssz*Bsz*Zsz,
                                         z_bf, kld, 1, 0);
  gemm(z_bf, phi_z_w_b, phi_z_b, nullptr, dech, Bsz, Hsz, Zsz, 2*Hsz, 0);
  copy_h_to_dech<<<BH / 256, 256, 0, stream>>>(h_bf, dech);
  gemm(dech, dec_w_b, dec_b, nullptr, dec_bf, Bsz, Hsz, 2*Hsz, Hsz, 0);
  gemm(dec_bf, act_w_b, act_b, (float*)d_out, nullptr, Bsz, Asz, Hsz, Asz, 0);
  write_kld<<<1, 1, 0, stream>>>(kld, (float*)d_out);
}
